// MultiHeadRandomFeatureAttention_87024627351710
// MI455X (gfx1250) — matrix operand order repaired and checked by reading, not compiled, not run on hardware
//
#include <hip/hip_runtime.h>
#include <hip/hip_bf16.h>
#include <math.h>

// ---------------------------------------------------------------------------
// MultiHeadRandomFeatureAttention for MI455X (gfx1250, wave32, WMMA bf16)
// B=4, L=4096, DIM=1024, H=16, DH=64, NF=64
// ---------------------------------------------------------------------------

#define DIMC   1024
#define LEN    4096
#define BATCH  4
#define HEADS  16
#define DH     64
#define NF     64
#define ROWS   (BATCH * LEN)          // 16384
#define EPSC   1e-6f

typedef __attribute__((ext_vector_type(16))) __bf16 v16bf;
typedef __attribute__((ext_vector_type(8)))  __bf16 v8bf;
typedef __attribute__((ext_vector_type(8)))  float  v8f;

// ---------------- scalar bf16 helpers (RNE) --------------------------------
__device__ __forceinline__ unsigned short f2bf(float f) {
    unsigned u = __float_as_uint(f);
    return (unsigned short)((u + 0x7FFFu + ((u >> 16) & 1u)) >> 16);
}
__device__ __forceinline__ float bf2f(unsigned short h) {
    return __uint_as_float(((unsigned)h) << 16);
}
// ordered-float <-> uint encoding for atomicMax on floats (incl. negatives)
__device__ __forceinline__ unsigned encf(float x) {
    unsigned u = __float_as_uint(x);
    return (u & 0x80000000u) ? ~u : (u | 0x80000000u);
}
__device__ __forceinline__ float decf(unsigned e) {
    return __uint_as_float((e & 0x80000000u) ? (e ^ 0x80000000u) : ~e);
}

// ---------------- WMMA fragment loaders (CDNA5 wave32 layouts) -------------
// A (16x32 bf16): lanes 0-15 hold row M=lane, K {0..7,16..23};
//                 lanes 16-31 hold row M=lane-16, K {8..15,24..31}.
__device__ __forceinline__ v16bf load_a16(const unsigned short* src, int ld) {
    int lane = threadIdx.x & 31;
    int half = lane >> 4;
    const unsigned short* p = src + (size_t)(lane & 15) * ld + (half ? 8 : 0);
    union { v16bf v; v8bf h[2]; } u;
    u.h[0] = *(const v8bf*)(p);        // K k0+0..7   (or +8..15)
    u.h[1] = *(const v8bf*)(p + 16);   // K k0+16..23 (or +24..31)
    return u.v;
}
// B (32x16 bf16) loaded from an N-major (transposed) matrix: row n holds
// contiguous K. lanes 0-15: col N=lane, K {0..7,16..23}; lanes 16-31: K {8..15,24..31}.
__device__ __forceinline__ v16bf load_b16(const unsigned short* srcT, int ld) {
    int lane = threadIdx.x & 31;
    int half = lane >> 4;
    const unsigned short* p = srcT + (size_t)(lane & 15) * ld + (half ? 8 : 0);
    union { v16bf v; v8bf h[2]; } u;
    u.h[0] = *(const v8bf*)(p);
    u.h[1] = *(const v8bf*)(p + 16);
    return u.v;
}
__device__ __forceinline__ v8f wmma_bf16(v16bf a, v16bf b, v8f c) {
    return __builtin_amdgcn_wmma_f32_16x16x32_bf16(false, a, false, b,
                                                   (short)0, c, false, false);
}

// ---------------- utility kernels ------------------------------------------
__global__ void cvt_f32_bf16_k(const float* __restrict__ s,
                               unsigned short* __restrict__ d, int n) {
    for (int i = blockIdx.x * blockDim.x + threadIdx.x; i < n;
         i += gridDim.x * blockDim.x)
        d[i] = f2bf(s[i]);
}

// Wrf [1,H,DH,NF] f32 -> W0T [H][NF][DH] bf16 (N-major so B-frag loads are
// contiguous in K)
__global__ void cvt_w0t_k(const float* __restrict__ s,
                          unsigned short* __restrict__ d) {
    int i = blockIdx.x * blockDim.x + threadIdx.x;
    if (i >= HEADS * DH * NF) return;
    int h = i / (DH * NF);
    int r = i % (DH * NF);
    int dd = r / NF, f = r % NF;
    d[(h * NF + f) * DH + dd] = f2bf(s[i]);
}

__global__ void zero_u32_k(unsigned* __restrict__ p, int n) {
    for (int i = blockIdx.x * blockDim.x + threadIdx.x; i < n;
         i += gridDim.x * blockDim.x)
        p[i] = 0u;
}

// ---------------- 32x64 GEMM tile body (unroll-8, SSA fragments) -----------
// Computes a 32x64 output tile of  dst = src(bf16) @ W(bf16)^T  with K=1024.
// accA = rows m0..m0+15, accB = rows m0+16..m0+31, columns c0..c0+63.
// Fragments are fresh SSA values inside an unroll-8 window so the scheduler
// hoists global loads over earlier WMMAs without register copies or tail
// branches (partial s_wait_loadcnt instead of full drains, no v_mov churn).
__device__ __forceinline__ void gemm32x64_u(
    const unsigned short* __restrict__ src, const unsigned short* __restrict__ W,
    int m0, int c0, v8f accA[4], v8f accB[4]) {
    const unsigned short* Ar0 = src + (size_t)m0 * DIMC;
    const unsigned short* Ar1 = Ar0 + (size_t)16 * DIMC;
    const unsigned short* Wc = W + (size_t)c0 * DIMC;

#pragma unroll 8
    for (int k0 = 0; k0 < DIMC; k0 += 32) {
        v16bf a0 = load_a16(Ar0 + k0, DIMC);
        v16bf a1 = load_a16(Ar1 + k0, DIMC);
        v16bf b0 = load_b16(Wc + k0, DIMC);
        v16bf b1 = load_b16(Wc + (size_t)16 * DIMC + k0, DIMC);
        v16bf b2 = load_b16(Wc + (size_t)32 * DIMC + k0, DIMC);
        v16bf b3 = load_b16(Wc + (size_t)48 * DIMC + k0, DIMC);
        accA[0] = wmma_bf16(a0, b0, accA[0]);
        accA[1] = wmma_bf16(a0, b1, accA[1]);
        accA[2] = wmma_bf16(a0, b2, accA[2]);
        accA[3] = wmma_bf16(a0, b3, accA[3]);
        accB[0] = wmma_bf16(a1, b0, accB[0]);
        accB[1] = wmma_bf16(a1, b1, accB[1]);
        accB[2] = wmma_bf16(a1, b2, accB[2]);
        accB[3] = wmma_bf16(a1, b3, accB[3]);
    }
}

// ---------------- K1: QKV projections (WMMA) -------------------------------
// y = x @ W^T + b ; write bf16 q/k/v; for k also accumulate sum(k^2) per row.
// Each wave computes a 32x64 output tile. grid.z selects q/k/v.
__global__ __launch_bounds__(256) void proj_qkv_k(
    const unsigned short* __restrict__ xb,
    const unsigned short* __restrict__ Wqb,
    const unsigned short* __restrict__ Wkb,
    const unsigned short* __restrict__ Wvb,
    const float* __restrict__ bq, const float* __restrict__ bk,
    const float* __restrict__ bv,
    unsigned short* __restrict__ qb, unsigned short* __restrict__ kb,
    unsigned short* __restrict__ vb, float* __restrict__ knorm) {
    int z = blockIdx.z;
    const unsigned short* W = (z == 0) ? Wqb : (z == 1) ? Wkb : Wvb;
    const float* bias = (z == 0) ? bq : (z == 1) ? bk : bv;
    unsigned short* dst = (z == 0) ? qb : (z == 1) ? kb : vb;

    int wave = threadIdx.x >> 5, lane = threadIdx.x & 31;
    int half = lane >> 4, ln = lane & 15;
    int task = blockIdx.x * 8 + wave;          // 8192 tasks (512 mt x 16 ns)
    int mt = task >> 4, ns = task & 15;
    int m0 = mt * 32, c0 = ns * 64;

    v8f accA[4], accB[4];
#pragma unroll
    for (int t = 0; t < 4; t++)
#pragma unroll
        for (int i = 0; i < 8; i++) { accA[t][i] = 0.0f; accB[t][i] = 0.0f; }

    gemm32x64_u(xb, W, m0, c0, accA, accB);

    float ssA[8], ssB[8];
#pragma unroll
    for (int i = 0; i < 8; i++) { ssA[i] = 0.0f; ssB[i] = 0.0f; }

#pragma unroll
    for (int t = 0; t < 4; t++) {
        float bval = bias[c0 + t * 16 + ln];
#pragma unroll
        for (int i = 0; i < 8; i++) {
            float va = accA[t][i] + bval;
            float vbv = accB[t][i] + bval;
            int ra = m0 + i + (half ? 8 : 0);
            int rb = ra + 16;
            dst[(size_t)ra * DIMC + c0 + t * 16 + ln] = f2bf(va);
            dst[(size_t)rb * DIMC + c0 + t * 16 + ln] = f2bf(vbv);
            ssA[i] += va * va;
            ssB[i] += vbv * vbv;
        }
    }
    if (z == 1) {  // k: accumulate sum-of-squares per (b,h,l)
#pragma unroll
        for (int i = 0; i < 8; i++) {
            float sa = ssA[i], sb = ssB[i];
            sa += __shfl_xor(sa, 1); sb += __shfl_xor(sb, 1);
            sa += __shfl_xor(sa, 2); sb += __shfl_xor(sb, 2);
            sa += __shfl_xor(sa, 4); sb += __shfl_xor(sb, 4);
            sa += __shfl_xor(sa, 8); sb += __shfl_xor(sb, 8);
            if (ln == 0) {
                int ra = m0 + i + (half ? 8 : 0);
                int rb = ra + 16;
                atomicAdd(&knorm[(((ra >> 12) * HEADS + ns) * LEN) + (ra & (LEN - 1))], sa);
                atomicAdd(&knorm[(((rb >> 12) * HEADS + ns) * LEN) + (rb & (LEN - 1))], sb);
            }
        }
    }
}

// ---------------- small K=64 WMMA body: 16 rows x 64 cols ------------------
// All fragments loaded up-front (2 A + 8 B), then a clean burst of 8 WMMAs.
__device__ __forceinline__ void rf_logits16x64(
    const unsigned short* __restrict__ rows,   // points at (rowbase, head k0=0)
    const unsigned short* __restrict__ w0h,    // W0T for this head [NF][DH]
    v8f acc[4]) {
    v16bf a0 = load_a16(rows, DIMC);
    v16bf a1 = load_a16(rows + 32, DIMC);
    v16bf b0[4], b1[4];
#pragma unroll
    for (int t = 0; t < 4; t++) {
        b0[t] = load_b16(w0h + (size_t)(t * 16) * DH, DH);
        b1[t] = load_b16(w0h + (size_t)(t * 16) * DH + 32, DH);
    }
#pragma unroll
    for (int t = 0; t < 4; t++) acc[t] = wmma_bf16(a0, b0[t], acc[t]);
#pragma unroll
    for (int t = 0; t < 4; t++) acc[t] = wmma_bf16(a1, b1[t], acc[t]);
}

// ---------------- K2: logits max pass (WMMA) -------------------------------
// logits = k @ W0 - 0.5*||k||^2 ; global max over (l,f) per (b,h).
__global__ __launch_bounds__(256) void logits_max_k(
    const unsigned short* __restrict__ kb,
    const unsigned short* __restrict__ w0t,
    const float* __restrict__ knorm, unsigned* __restrict__ maxenc) {
    int wave = threadIdx.x >> 5, lane = threadIdx.x & 31;
    int half = lane >> 4;
    int task = blockIdx.x * 8 + wave;          // 64 * 256 tasks
    int bh = task >> 8, lt = task & 255;
    int b = bh >> 4, h = bh & 15;
    int l0 = lt * 16;
    int rowbase = b * LEN + l0;

    v8f acc[4];
#pragma unroll
    for (int t = 0; t < 4; t++)
#pragma unroll
        for (int i = 0; i < 8; i++) acc[t][i] = 0.0f;

    rf_logits16x64(kb + (size_t)rowbase * DIMC + h * DH, w0t + (size_t)h * NF * DH, acc);

    float m = -3.0e38f;
#pragma unroll
    for (int i = 0; i < 8; i++) {
        int l = l0 + i + (half ? 8 : 0);
        float kn = 0.5f * knorm[bh * LEN + l];
#pragma unroll
        for (int t = 0; t < 4; t++) m = fmaxf(m, acc[t][i] - kn);
    }
    m = fmaxf(m, __shfl_xor(m, 1));
    m = fmaxf(m, __shfl_xor(m, 2));
    m = fmaxf(m, __shfl_xor(m, 4));
    m = fmaxf(m, __shfl_xor(m, 8));
    m = fmaxf(m, __shfl_xor(m, 16));
    if (lane == 0) atomicMax(&maxenc[bh], encf(m));
}

// ---------------- K3: exp pass, write unnormalized alpha (bf16) ------------
__global__ __launch_bounds__(256) void logits_exp_k(
    const unsigned short* __restrict__ kb,
    const unsigned short* __restrict__ w0t,
    const float* __restrict__ knorm, const unsigned* __restrict__ maxenc,
    unsigned short* __restrict__ alpha) {
    int wave = threadIdx.x >> 5, lane = threadIdx.x & 31;
    int half = lane >> 4, ln = lane & 15;
    int task = blockIdx.x * 8 + wave;
    int bh = task >> 8, lt = task & 255;
    int b = bh >> 4, h = bh & 15;
    int l0 = lt * 16;
    int rowbase = b * LEN + l0;
    float mval = decf(maxenc[bh]);

    v8f acc[4];
#pragma unroll
    for (int t = 0; t < 4; t++)
#pragma unroll
        for (int i = 0; i < 8; i++) acc[t][i] = 0.0f;

    rf_logits16x64(kb + (size_t)rowbase * DIMC + h * DH, w0t + (size_t)h * NF * DH, acc);

#pragma unroll
    for (int i = 0; i < 8; i++) {
        int l = l0 + i + (half ? 8 : 0);
        float kn = 0.5f * knorm[bh * LEN + l];
#pragma unroll
        for (int t = 0; t < 4; t++) {
            float e = __expf(acc[t][i] - kn - mval);
            alpha[((size_t)(bh * LEN + l)) * NF + t * 16 + ln] = f2bf(e);
        }
    }
}

// ---------------- K4: Ntilde = alpha^T @ v, Dtilde = sum_l alpha -----------
// LDS-staged VALU accumulation; blocks = (b,h) x 8 l-chunks of 512.
__global__ __launch_bounds__(256) void accum_nd_k(
    const unsigned short* __restrict__ alpha,
    const unsigned short* __restrict__ vb, float* __restrict__ Nacc,
    float* __restrict__ Dacc) {
    __shared__ float Als[64 * 64];
    __shared__ float Vls[64 * 64];
    int t = threadIdx.x;
    int bh = blockIdx.x >> 3;
    int chunk = blockIdx.x & 7;
    int b = bh >> 4, h = bh & 15;
    int d = t & 63, fg = (t >> 6) * 16;

    float accv[16];
#pragma unroll
    for (int j = 0; j < 16; j++) accv[j] = 0.0f;
    float dsum = 0.0f;

    int lbase = chunk * 512;
    for (int lb = 0; lb < 512; lb += 64) {
        for (int i = t; i < 4096; i += 256) {
            int lr = i >> 6, c = i & 63;
            int l = lbase + lb + lr;
            Als[i] = bf2f(alpha[((size_t)(bh * LEN + l)) * NF + c]);
            Vls[i] = bf2f(vb[((size_t)(b * LEN + l)) * DIMC + h * DH + c]);
        }
        __syncthreads();
        for (int lr = 0; lr < 64; lr++) {
            float vv = Vls[lr * 64 + d];
#pragma unroll
            for (int j = 0; j < 16; j++) accv[j] += Als[lr * 64 + fg + j] * vv;
        }
        if (t < 64)
            for (int lr = 0; lr < 64; lr++) dsum += Als[lr * 64 + t];
        __syncthreads();
    }
#pragma unroll
    for (int j = 0; j < 16; j++)
        atomicAdd(&Nacc[(size_t)bh * NF * DH + (fg + j) * DH + d], accv[j]);
    if (t < 64) atomicAdd(&Dacc[bh * NF + t], dsum);
}

// ---------------- K4b: Z = sum_f Dtilde ------------------------------------
__global__ void reduce_z_k(const float* __restrict__ Dacc,
                           float* __restrict__ Z) {
    int t = threadIdx.x;
    if (t < BATCH * HEADS) {
        float s = 0.0f;
        for (int f = 0; f < NF; f++) s += Dacc[t * NF + f];
        Z[t] = s;
    }
}

// ---------------- K5: phi softmax + num/den -> ctx (bf16) ------------------
// phi logits via WMMA; per-row softmax via shuffles; num/den via LDS VALU.
__global__ __launch_bounds__(256) void phi_out_k(
    const unsigned short* __restrict__ qb,
    const unsigned short* __restrict__ w0t, const float* __restrict__ Nacc,
    const float* __restrict__ Dacc, const float* __restrict__ Z,
    unsigned short* __restrict__ ctx) {
    __shared__ float Nls[NF * DH];
    __shared__ float Dls[NF];
    __shared__ float Phils[8 * 16 * 64];
    __shared__ float Zs;

    int t = threadIdx.x, wave = t >> 5, lane = t & 31;
    int half = lane >> 4, ln = lane & 15;
    int bh = blockIdx.x >> 5;          // 32 chunks of 128 rows
    int chunk = blockIdx.x & 31;
    int b = bh >> 4, h = bh & 15;

    for (int i = t; i < NF * DH; i += 256) Nls[i] = Nacc[(size_t)bh * NF * DH + i];
    if (t < NF) Dls[t] = Dacc[bh * NF + t];
    if (t == 0) Zs = Z[bh];
    __syncthreads();

    int l0 = chunk * 128 + wave * 16;
    int rowbase = b * LEN + l0;

    v8f acc[4];
#pragma unroll
    for (int tt = 0; tt < 4; tt++)
#pragma unroll
        for (int i = 0; i < 8; i++) acc[tt][i] = 0.0f;

    rf_logits16x64(qb + (size_t)rowbase * DIMC + h * DH, w0t + (size_t)h * NF * DH, acc);

    float* myphi = &Phils[wave * 16 * 64];
#pragma unroll
    for (int i = 0; i < 8; i++) {
        float mx = fmaxf(fmaxf(acc[0][i], acc[1][i]),
                         fmaxf(acc[2][i], acc[3][i]));
        mx = fmaxf(mx, __shfl_xor(mx, 1));
        mx = fmaxf(mx, __shfl_xor(mx, 2));
        mx = fmaxf(mx, __shfl_xor(mx, 4));
        mx = fmaxf(mx, __shfl_xor(mx, 8));
        float e[4];
        float s = 0.0f;
#pragma unroll
        for (int tt = 0; tt < 4; tt++) {
            e[tt] = __expf(acc[tt][i] - mx);
            s += e[tt];
        }
        s += __shfl_xor(s, 1);
        s += __shfl_xor(s, 2);
        s += __shfl_xor(s, 4);
        s += __shfl_xor(s, 8);
        float inv = 1.0f / s;
        int rloc = i + (half ? 8 : 0);
#pragma unroll
        for (int tt = 0; tt < 4; tt++)
            myphi[rloc * 64 + tt * 16 + ln] = e[tt] * inv;
    }
    // same-wave LDS store->load: LDS ops are in-order within a wave.

    float zeps = EPSC * Zs;
    int d0 = lane * 2;
    for (int r = 0; r < 16; r++) {
        float n0 = 0.0f, n1 = 0.0f, den = 0.0f;
        for (int f = 0; f < NF; f++) {
            float p = myphi[r * 64 + f];
            den += p * Dls[f];
            n0 += p * Nls[f * DH + d0];
            n1 += p * Nls[f * DH + d0 + 1];
        }
        float rinv = 1.0f / (den + zeps);
        int l = l0 + r;
        unsigned pk = ((unsigned)f2bf(n1 * rinv) << 16) | (unsigned)f2bf(n0 * rinv);
        *(unsigned*)(ctx + ((size_t)(b * LEN + l)) * DIMC + h * DH + d0) = pk;
    }
}

// ---------------- K6: output projection (WMMA, 32x64 tiles) ----------------
__global__ __launch_bounds__(256) void proj_out_k(
    const unsigned short* __restrict__ ctxb,
    const unsigned short* __restrict__ Wob, const float* __restrict__ bo,
    float* __restrict__ out) {
    int wave = threadIdx.x >> 5, lane = threadIdx.x & 31;
    int half = lane >> 4, ln = lane & 15;
    int task = blockIdx.x * 8 + wave;          // 8192 tasks
    int mt = task >> 4, ns = task & 15;
    int m0 = mt * 32, c0 = ns * 64;

    v8f accA[4], accB[4];
#pragma unroll
    for (int t = 0; t < 4; t++)
#pragma unroll
        for (int i = 0; i < 8; i++) { accA[t][i] = 0.0f; accB[t][i] = 0.0f; }

    gemm32x64_u(ctxb, Wob, m0, c0, accA, accB);

#pragma unroll
    for (int t = 0; t < 4; t++) {
        float bval = bo[c0 + t * 16 + ln];
#pragma unroll
        for (int i = 0; i < 8; i++) {
            int ra = m0 + i + (half ? 8 : 0);
            int rb = ra + 16;
            out[(size_t)ra * DIMC + c0 + t * 16 + ln] = accA[t][i] + bval;
            out[(size_t)rb * DIMC + c0 + t * 16 + ln] = accB[t][i] + bval;
        }
    }
}

// ---------------------------------------------------------------------------
extern "C" void kernel_launch(void* const* d_in, const int* in_sizes, int n_in,
                              void* d_out, int out_size, void* d_ws,
                              size_t ws_size, hipStream_t stream) {
    const float* x   = (const float*)d_in[0];
    const float* Wq  = (const float*)d_in[1];
    const float* bq  = (const float*)d_in[2];
    const float* Wk  = (const float*)d_in[3];
    const float* bk  = (const float*)d_in[4];
    const float* Wv  = (const float*)d_in[5];
    const float* bv  = (const float*)d_in[6];
    const float* Wo  = (const float*)d_in[7];
    const float* bo  = (const float*)d_in[8];
    const float* Wrf = (const float*)d_in[9];
    float* out = (float*)d_out;

    const size_t NX = (size_t)ROWS * DIMC;      // 16,777,216
    const size_t NW = (size_t)DIMC * DIMC;      // 1,048,576

    char* ws = (char*)d_ws;
    size_t off = 0;
    auto take = [&](size_t bytes) -> char* {
        char* p = ws + off;
        off = (off + bytes + 255) & ~(size_t)255;
        return p;
    };
    unsigned short* xb    = (unsigned short*)take(NX * 2);
    unsigned short* Wqb   = (unsigned short*)take(NW * 2);
    unsigned short* Wkb   = (unsigned short*)take(NW * 2);
    unsigned short* Wvb   = (unsigned short*)take(NW * 2);
    unsigned short* Wob   = (unsigned short*)take(NW * 2);
    unsigned short* w0t   = (unsigned short*)take((size_t)HEADS * NF * DH * 2);
    unsigned short* qb    = (unsigned short*)take(NX * 2);
    unsigned short* kb    = (unsigned short*)take(NX * 2);
    unsigned short* vb    = (unsigned short*)take(NX * 2);
    unsigned short* alpha = (unsigned short*)take((size_t)BATCH * HEADS * LEN * NF * 2);
    unsigned short* ctx   = (unsigned short*)take(NX * 2);
    // zero-initialized stats region (contiguous)
    char* zbase = ws + off;
    float*    knorm  = (float*)take((size_t)BATCH * HEADS * LEN * 4);
    unsigned* maxenc = (unsigned*)take(BATCH * HEADS * 4);
    float*    Nacc   = (float*)take((size_t)BATCH * HEADS * NF * DH * 4);
    float*    Dacc   = (float*)take((size_t)BATCH * HEADS * NF * 4);
    float*    Zbuf   = (float*)take(BATCH * HEADS * 4);
    size_t zwords = ((ws + off) - zbase) / 4;

    // 0) conversions
    cvt_f32_bf16_k<<<2048, 256, 0, stream>>>(x,  xb,  (int)NX);
    cvt_f32_bf16_k<<<1024, 256, 0, stream>>>(Wq, Wqb, (int)NW);
    cvt_f32_bf16_k<<<1024, 256, 0, stream>>>(Wk, Wkb, (int)NW);
    cvt_f32_bf16_k<<<1024, 256, 0, stream>>>(Wv, Wvb, (int)NW);
    cvt_f32_bf16_k<<<1024, 256, 0, stream>>>(Wo, Wob, (int)NW);
    cvt_w0t_k<<<(HEADS * DH * NF + 255) / 256, 256, 0, stream>>>(Wrf, w0t);
    zero_u32_k<<<512, 256, 0, stream>>>((unsigned*)zbase, (int)zwords);

    // 1) QKV projections (WMMA, 32x64 tiles) + k row norms
    proj_qkv_k<<<dim3(1024, 1, 3), 256, 0, stream>>>(
        xb, Wqb, Wkb, Wvb, bq, bk, bv, qb, kb, vb, knorm);

    // 2) logits global max
    logits_max_k<<<2048, 256, 0, stream>>>(kb, w0t, knorm, maxenc);

    // 3) exp pass -> unnormalized alpha
    logits_exp_k<<<2048, 256, 0, stream>>>(kb, w0t, knorm, maxenc, alpha);

    // 4) Ntilde / Dtilde
    accum_nd_k<<<BATCH * HEADS * 8, 256, 0, stream>>>(alpha, vb, Nacc, Dacc);
    reduce_z_k<<<1, 64, 0, stream>>>(Dacc, Zbuf);

    // 5) phi + num/den -> ctx
    phi_out_k<<<BATCH * HEADS * 32, 256, 0, stream>>>(qb, w0t, Nacc, Dacc,
                                                      Zbuf, ctx);

    // 6) output projection (WMMA, 32x64 tiles)
    proj_out_k<<<1024, 256, 0, stream>>>(ctx, Wob, bo, out);
}
